// SelfAttention_39840116638576
// MI455X (gfx1250) — matrix operand order repaired and checked by reading, not compiled, not run on hardware
//
#include <hip/hip_runtime.h>
#include <hip/hip_bf16.h>

#define S_LEN   4096
#define D_MODEL 1024

typedef __attribute__((ext_vector_type(16))) __bf16       v16bf;
typedef __attribute__((ext_vector_type(8)))  float        v8f;
typedef __attribute__((ext_vector_type(4)))  float        v4f;
typedef __attribute__((ext_vector_type(4)))  unsigned int v4u;
typedef __attribute__((ext_vector_type(2)))  unsigned int v2u;

// ---------------------------------------------------------------------------
// Fragment layouts (CDNA5 ISA §7.12.2, wave32):
//  A 16x32 bf16: lane L -> row M=L%16; elems 0..7  = K[(L/16)*8 .. +7],
//                elems 8..15 = same +16.   Two 16B loads.
//  B 32x16 bf16 (from Bt[N][K] row-major): lane L -> row N=L%16,
//                elems 0..7 = K[(L/16)*8 .. +7], elems 8..15 = same +16. Two 16B loads.
// Works identically for global and LDS source pointers.
// ---------------------------------------------------------------------------
union FragU { v4u u[2]; v16bf v; };

__device__ __forceinline__ v16bf load_frag_a(const __bf16* p) {
  FragU t;
  t.u[0] = *reinterpret_cast<const v4u*>(p);
  t.u[1] = *reinterpret_cast<const v4u*>(p + 16);
  return t.v;
}

__device__ __forceinline__ v16bf load_frag_b(const __bf16* p) {
  FragU t;
  t.u[0] = *reinterpret_cast<const v4u*>(p);
  t.u[1] = *reinterpret_cast<const v4u*>(p + 16);
  return t.v;
}

enum OutMode { OUT_F32 = 0, OUT_BF16 = 1, OUT_BF16_T = 2 };

// C/D layout: lane L, reg r -> (M = row0 + r, N = col); caller folds the
// +8-per-half-wave into row0.  OUT_BF16_T: 8 values/lane are contiguous
// (vT[n][m0..m0+7]) -> single 16B store.
template <int MODE>
__device__ __forceinline__ void store_tile(void* __restrict__ Cv, v8f acc,
                                           int row0, int col, int ldc,
                                           float scale) {
  if constexpr (MODE == OUT_F32) {
    float* C = reinterpret_cast<float*>(Cv);
#pragma unroll
    for (int r = 0; r < 8; ++r)
      C[(size_t)(row0 + r) * ldc + col] = acc[r] * scale;
  } else if constexpr (MODE == OUT_BF16) {
    __bf16* C = reinterpret_cast<__bf16*>(Cv);
#pragma unroll
    for (int r = 0; r < 8; ++r)
      C[(size_t)(row0 + r) * ldc + col] = (__bf16)(acc[r] * scale);
  } else {  // OUT_BF16_T
    __bf16* C = reinterpret_cast<__bf16*>(Cv);
    union { __bf16 h[8]; v4u u; } t;
#pragma unroll
    for (int r = 0; r < 8; ++r) t.h[r] = (__bf16)(acc[r] * scale);
    *reinterpret_cast<v4u*>(C + (size_t)col * ldc + row0) = t.u;
  }
}

// ---------------------------------------------------------------------------
// C[M,N] = scale * A[M,K] @ Bt[N,K]^T   (A, Bt row-major bf16, f32 accum)
//
// Block: 256 threads = 8 wave32s; block tile 256x128; wave tile 64x64
// (4x4 WMMA tiles -> 16 v_wmma per 32-wide k-step from 8 LDS fragments).
// Double-buffered LDS staging: per k-step the block pulls a 256x32 A tile
// and a 128x32 B tile (24 KB) from global/L2 -> ~87 FLOP per global byte;
// fragment bandwidth is served by LDS.  Rows padded to 40 bf16 (80 B =
// 20 banks); gcd(20,64)=4 with 4-bank-wide b128 accesses => the 16 lanes
// of each half-wave hit disjoint 4-bank groups (conflict-free).
// One barrier per k-step; global loads for stage k+1 are issued before the
// WMMAs of stage k so the s_wait_loadcnt lands at the ds_store afterwards.
// Grids divide exactly -> EXEC all 1s everywhere (WMMA requirement).
// ---------------------------------------------------------------------------
template <int MODE>
__global__ __launch_bounds__(256) void gemm_abt(
    const __bf16* __restrict__ A, const __bf16* __restrict__ B,
    void* __restrict__ Cv, int K, int lda, int ldb, int ldc, float scale) {
  constexpr int BM = 256, BN = 128;
  constexpr int PITCH = 40;  // 32 + 8 pad, conflict-free, keeps 16B alignment
  __shared__ __bf16 As[2][BM * PITCH];  // 40 KB
  __shared__ __bf16 Bs[2][BN * PITCH];  // 20 KB

  const int bm   = blockIdx.x * BM;
  const int bn   = blockIdx.y * BN;
  const int t    = threadIdx.x;
  const int w    = t >> 5;           // wave id (wave32)
  const int lane = t & 31;
  const int wm   = (w & 3) * 64;     // 4 waves across M
  const int wn   = (w >> 2) * 64;    // 2 waves across N
  const int kh   = lane >> 4;
  const int l16  = lane & 15;

  // Staging: A tile 256x32 = 1024 16B-chunks (4/thread);
  //          B tile 128x32 =  512 16B-chunks (2/thread).
  const __bf16* gA[4];
  const __bf16* gB[2];
  int sA[4], sB[2];
#pragma unroll
  for (int i = 0; i < 4; ++i) {
    const int c = t + i * 256, r = c >> 2, cc = (c & 3) * 8;
    gA[i] = A + (size_t)(bm + r) * lda + cc;
    sA[i] = r * PITCH + cc;
  }
#pragma unroll
  for (int i = 0; i < 2; ++i) {
    const int c = t + i * 256, r = c >> 2, cc = (c & 3) * 8;
    gB[i] = B + (size_t)(bn + r) * ldb + cc;
    sB[i] = r * PITCH + cc;
  }

  v8f acc[4][4];
#pragma unroll
  for (int mt = 0; mt < 4; ++mt)
#pragma unroll
    for (int nt = 0; nt < 4; ++nt)
      acc[mt][nt] = v8f{0.f, 0.f, 0.f, 0.f, 0.f, 0.f, 0.f, 0.f};

  v4u ra[4], rb[2];
  auto ldg = [&](int kk) {
    const size_t o = (size_t)kk * 32;
#pragma unroll
    for (int i = 0; i < 4; ++i)
      ra[i] = *reinterpret_cast<const v4u*>(gA[i] + o);
#pragma unroll
    for (int i = 0; i < 2; ++i)
      rb[i] = *reinterpret_cast<const v4u*>(gB[i] + o);
  };
  auto stl = [&](int buf) {
#pragma unroll
    for (int i = 0; i < 4; ++i)
      *reinterpret_cast<v4u*>(&As[buf][sA[i]]) = ra[i];
#pragma unroll
    for (int i = 0; i < 2; ++i)
      *reinterpret_cast<v4u*>(&Bs[buf][sB[i]]) = rb[i];
  };

  ldg(0);
  stl(0);
  __syncthreads();

  const int nk = K >> 5;
  for (int kk = 0; kk < nk; ++kk) {
    const int  buf  = kk & 1;
    const bool more = (kk + 1) < nk;
    if (more) ldg(kk + 1);  // prefetch next stage into registers

    v16bf afr[4];
#pragma unroll
    for (int mt = 0; mt < 4; ++mt)
      afr[mt] = load_frag_a(&As[buf][(wm + mt * 16 + l16) * PITCH + kh * 8]);

#pragma unroll
    for (int nt = 0; nt < 4; ++nt) {
      v16bf bfr =
          load_frag_b(&Bs[buf][(wn + nt * 16 + l16) * PITCH + kh * 8]);
#pragma unroll
      for (int mt = 0; mt < 4; ++mt)
        acc[mt][nt] = __builtin_amdgcn_wmma_f32_16x16x32_bf16(
            false, afr[mt], false, bfr, (short)0, acc[mt][nt], false, false);
    }

    if (more) stl(buf ^ 1);  // safe: last readers of buf^1 finished before
                             // the barrier that ended iteration kk-1
    __syncthreads();
  }

  const int rr = bm + wm + 8 * kh;
  const int cc = bn + wn + l16;
#pragma unroll
  for (int mt = 0; mt < 4; ++mt)
#pragma unroll
    for (int nt = 0; nt < 4; ++nt)
      store_tile<MODE>(Cv, acc[mt][nt], rr + mt * 16, cc + nt * 16, ldc,
                       scale);
}

// ---------------------------------------------------------------------------
// f32 -> bf16 convert, 4 elements/thread
// ---------------------------------------------------------------------------
__global__ __launch_bounds__(256) void cvt_f32_bf16(
    const float* __restrict__ in, __bf16* __restrict__ out, int n4) {
  int i = blockIdx.x * blockDim.x + threadIdx.x;
  if (i >= n4) return;
  v4f f = reinterpret_cast<const v4f*>(in)[i];
  union { __bf16 h[4]; v2u u; } t;
  t.h[0] = (__bf16)f[0];
  t.h[1] = (__bf16)f[1];
  t.h[2] = (__bf16)f[2];
  t.h[3] = (__bf16)f[3];
  reinterpret_cast<v2u*>(out)[i] = t.u;
}

// ---------------------------------------------------------------------------
// Row softmax over p[S,S] in place (f32 p_attn output) + bf16 copy for the
// PV matmul.  One 256-thread block per row; 16 elements/thread in registers
// -> single global read of the row.
// ---------------------------------------------------------------------------
__global__ __launch_bounds__(256) void softmax_rows(
    float* __restrict__ p, __bf16* __restrict__ pbf) {
  __shared__ float red[256];
  const int    tid = threadIdx.x;
  const size_t row = blockIdx.x;
  float*  pr  = p   + row * S_LEN;
  __bf16* pbr = pbf + row * S_LEN;

  float v[16];
  float m = -INFINITY;
#pragma unroll
  for (int j = 0; j < 16; ++j) {
    v[j] = pr[tid + j * 256];
    m = fmaxf(m, v[j]);
  }
  red[tid] = m;
  __syncthreads();
  for (int st = 128; st > 0; st >>= 1) {
    if (tid < st) red[tid] = fmaxf(red[tid], red[tid + st]);
    __syncthreads();
  }
  m = red[0];
  __syncthreads();

  float s = 0.f;
#pragma unroll
  for (int j = 0; j < 16; ++j) {
    v[j] = __expf(v[j] - m);
    s += v[j];
  }
  red[tid] = s;
  __syncthreads();
  for (int st = 128; st > 0; st >>= 1) {
    if (tid < st) red[tid] += red[tid + st];
    __syncthreads();
  }
  const float inv = 1.0f / red[0];

#pragma unroll
  for (int j = 0; j < 16; ++j) {
    float e = v[j] * inv;
    pr[tid + j * 256]  = e;
    pbr[tid + j * 256] = (__bf16)e;
  }
}

// ---------------------------------------------------------------------------
// Orchestration.  d_out = [ Vs (S*D f32) | p_attn (S*S f32) ].
// Workspace (70 MiB): x_bf 8M | Wq/Wk/Wv_bf 2M each | q_bf 8M | k_bf 8M |
//                     vT_bf 8M | p_bf 32M
// ---------------------------------------------------------------------------
extern "C" void kernel_launch(void* const* d_in, const int* in_sizes, int n_in,
                              void* d_out, int out_size, void* d_ws,
                              size_t ws_size, hipStream_t stream) {
  (void)in_sizes; (void)n_in; (void)out_size; (void)ws_size;

  const float* x  = (const float*)d_in[0];
  const float* Wq = (const float*)d_in[1];
  const float* Wk = (const float*)d_in[2];
  const float* Wv = (const float*)d_in[3];

  float* Vs = (float*)d_out;
  float* P  = Vs + (size_t)S_LEN * D_MODEL;

  char*  ws  = (char*)d_ws;
  size_t off = 0;
  auto carve = [&](size_t bytes) { void* p = ws + off; off += bytes; return p; };
  const size_t SD = (size_t)S_LEN * D_MODEL;
  const size_t DD = (size_t)D_MODEL * D_MODEL;
  const size_t SS = (size_t)S_LEN * S_LEN;

  __bf16* x_bf  = (__bf16*)carve(SD * 2);
  __bf16* Wq_bf = (__bf16*)carve(DD * 2);
  __bf16* Wk_bf = (__bf16*)carve(DD * 2);
  __bf16* Wv_bf = (__bf16*)carve(DD * 2);
  __bf16* q_bf  = (__bf16*)carve(SD * 2);
  __bf16* k_bf  = (__bf16*)carve(SD * 2);
  __bf16* vT_bf = (__bf16*)carve(SD * 2);  // [D_MODEL][S_LEN]
  __bf16* p_bf  = (__bf16*)carve(SS * 2);

  // 1) f32 -> bf16 inputs
  cvt_f32_bf16<<<(int)(SD / 4 / 256), 256, 0, stream>>>(x,  x_bf,  (int)(SD / 4));
  cvt_f32_bf16<<<(int)(DD / 4 / 256), 256, 0, stream>>>(Wq, Wq_bf, (int)(DD / 4));
  cvt_f32_bf16<<<(int)(DD / 4 / 256), 256, 0, stream>>>(Wk, Wk_bf, (int)(DD / 4));
  cvt_f32_bf16<<<(int)(DD / 4 / 256), 256, 0, stream>>>(Wv, Wv_bf, (int)(DD / 4));

  // 2) projections: q = x @ Wq^T, k = x @ Wk^T, vT = (x @ Wv^T)^T
  dim3 gProj(S_LEN / 256, D_MODEL / 128);
  gemm_abt<OUT_BF16><<<gProj, 256, 0, stream>>>(x_bf, Wq_bf, q_bf,
      D_MODEL, D_MODEL, D_MODEL, D_MODEL, 1.0f);
  gemm_abt<OUT_BF16><<<gProj, 256, 0, stream>>>(x_bf, Wk_bf, k_bf,
      D_MODEL, D_MODEL, D_MODEL, D_MODEL, 1.0f);
  gemm_abt<OUT_BF16_T><<<gProj, 256, 0, stream>>>(x_bf, Wv_bf, vT_bf,
      D_MODEL, D_MODEL, D_MODEL, S_LEN, 1.0f);  // ldc = S (vT row stride)

  // 3) scores = q @ k^T / sqrt(D)   (f32 into p_attn slice of d_out)
  dim3 gScore(S_LEN / 256, S_LEN / 128);
  gemm_abt<OUT_F32><<<gScore, 256, 0, stream>>>(q_bf, k_bf, P,
      D_MODEL, D_MODEL, D_MODEL, S_LEN, 0.03125f);  // 1/sqrt(1024)

  // 4) softmax rows (f32 in place) + bf16 copy for PV
  softmax_rows<<<S_LEN, 256, 0, stream>>>(P, p_bf);

  // 5) Vs = p @ v = p_bf @ vT_bf^T   (f32 into d_out)
  dim3 gAV(S_LEN / 256, D_MODEL / 128);
  gemm_abt<OUT_F32><<<gAV, 256, 0, stream>>>(p_bf, vT_bf, Vs,
      S_LEN, S_LEN, S_LEN, D_MODEL, 1.0f);
}
